// TemporalEncoder_84189948936273
// MI455X (gfx1250) — compile-verified, matrix operand order checked by reading, not run on hardware
//
#include <hip/hip_runtime.h>
#include <hip/hip_bf16.h>
#include <math.h>

// Problem constants (from reference): B=32, T=512, IN=2048, H=1024, OUT=2048
#define B_   32
#define T_   512
#define IN_  2048
#define H_   1024
#define OUT_ 2048
#define G4_  (4 * H_)

typedef __attribute__((ext_vector_type(16))) __bf16 v16bf;
typedef __attribute__((ext_vector_type(8)))  float  v8f;

// ---------------------------------------------------------------------------
// Pack a row-major f32 matrix [R, K] into CDNA5 WMMA bf16 tile layout.
// Tile = 16 rows x 32 K.  Layout: [rt][kt][lane][16 bf16], lane-contiguous so a
// lane fetches its whole fragment with one 32B load.
// Per ISA 16-bit A layout: lane L (0-15) = row L, holds K = {0..7, 16..23};
// lane L+16 = row L, holds K = {8..15, 24..31}; 2 bf16 per dword.
// Same layout serves the B operand (weights stored [N][K] row-major ==
// column-major 32x16 B tiles).
// ---------------------------------------------------------------------------
__global__ void pack_tiles_kernel(const float* __restrict__ src,
                                  __hip_bfloat16* __restrict__ dst,
                                  int R, int K) {
  const int Kt = K >> 5;
  long idx = (long)blockIdx.x * blockDim.x + threadIdx.x;  // one thread = one (tile, lane)
  long total = (long)(R >> 4) * Kt * 32;
  if (idx >= total) return;
  int  lane = (int)(idx & 31);
  long tile = idx >> 5;
  int  kt   = (int)(tile % Kt);
  long rt   = tile / Kt;
  int  r    = (int)(rt << 4) + (lane & 15);
  int  khi  = (lane >> 4) << 3;                      // hi lanes hold K+8 groups
  const float* in = src + (long)r * K + (kt << 5);
  __hip_bfloat16* out = dst + idx * 16;
#pragma unroll
  for (int v = 0; v < 8; ++v) {
    int kb = ((v & 3) << 1) + ((v >> 2) << 4) + khi; // dword v -> K base
    out[2 * v]     = __float2bfloat16(in[kb]);
    out[2 * v + 1] = __float2bfloat16(in[kb + 1]);
  }
}

// ---------------------------------------------------------------------------
// Generic bf16 WMMA GEMM:  C[M,N] f32 = Apk[M,K] * Bpk[N,K]^T (+ bias[n]).
// Both operands pre-packed.  Block = 256 thr = 8 waves, wave grid 2(M) x 4(N);
// wave tile = 32x64 -> 8 f32 accumulators.  Used for the large batched GEMMs.
// ---------------------------------------------------------------------------
__global__ void __launch_bounds__(256)
gemm_bf16_tn(const __hip_bfloat16* __restrict__ Apk,
             const __hip_bfloat16* __restrict__ Bpk,
             float* __restrict__ C,
             const float* __restrict__ bias,
             int M, int N, int K) {
  const int Kt    = K >> 5;
  const int lane  = threadIdx.x & 31;
  const int wave  = threadIdx.x >> 5;
  const int mBase = blockIdx.y * 64 + (wave >> 2) * 32;
  const int nBase = blockIdx.x * 256 + (wave & 3) * 64;
  if (mBase >= M) return;

  const v16bf* A  = (const v16bf*)Apk;
  const v16bf* Bm = (const v16bf*)Bpk;

  long aIdx[2], bIdx[4];
#pragma unroll
  for (int i = 0; i < 2; ++i) aIdx[i] = ((long)((mBase >> 4) + i) * Kt) * 32 + lane;
#pragma unroll
  for (int j = 0; j < 4; ++j) bIdx[j] = ((long)((nBase >> 4) + j) * Kt) * 32 + lane;

  v8f acc[2][4];
#pragma unroll
  for (int i = 0; i < 2; ++i)
#pragma unroll
    for (int j = 0; j < 4; ++j) acc[i][j] = {};

  for (int kt = 0; kt < Kt; ++kt) {
    v16bf a0 = A[aIdx[0]];
    v16bf a1 = A[aIdx[1]];
    v16bf b0 = Bm[bIdx[0]];
    v16bf b1 = Bm[bIdx[1]];
    v16bf b2 = Bm[bIdx[2]];
    v16bf b3 = Bm[bIdx[3]];
    acc[0][0] = __builtin_amdgcn_wmma_f32_16x16x32_bf16(false, a0, false, b0, (short)0, acc[0][0], false, false);
    acc[0][1] = __builtin_amdgcn_wmma_f32_16x16x32_bf16(false, a0, false, b1, (short)0, acc[0][1], false, false);
    acc[0][2] = __builtin_amdgcn_wmma_f32_16x16x32_bf16(false, a0, false, b2, (short)0, acc[0][2], false, false);
    acc[0][3] = __builtin_amdgcn_wmma_f32_16x16x32_bf16(false, a0, false, b3, (short)0, acc[0][3], false, false);
    acc[1][0] = __builtin_amdgcn_wmma_f32_16x16x32_bf16(false, a1, false, b0, (short)0, acc[1][0], false, false);
    acc[1][1] = __builtin_amdgcn_wmma_f32_16x16x32_bf16(false, a1, false, b1, (short)0, acc[1][1], false, false);
    acc[1][2] = __builtin_amdgcn_wmma_f32_16x16x32_bf16(false, a1, false, b2, (short)0, acc[1][2], false, false);
    acc[1][3] = __builtin_amdgcn_wmma_f32_16x16x32_bf16(false, a1, false, b3, (short)0, acc[1][3], false, false);
#pragma unroll
    for (int i = 0; i < 2; ++i) aIdx[i] += 32;
#pragma unroll
    for (int j = 0; j < 4; ++j) bIdx[j] += 32;
  }

  // D layout: VGPR v holds M = v (lanes 0-15) / M = v+8 (lanes 16-31), N = lane%16
  const int mrow = (lane >> 4) << 3;
  const int ncol = lane & 15;
#pragma unroll
  for (int i = 0; i < 2; ++i) {
#pragma unroll
    for (int j = 0; j < 4; ++j) {
      int gm = mBase + (i << 4) + mrow;
      int gn = nBase + (j << 4) + ncol;
      float bv = bias ? bias[gn] : 0.0f;
      float* cp = C + (long)gm * N + gn;
#pragma unroll
      for (int v = 0; v < 8; ++v) cp[(long)v * N] = acc[i][j][v] + bv;
    }
  }
}

// ---------------------------------------------------------------------------
// Fused LSTM recurrent step:  Grec = h_{t-1} @ Whh^T  (WMMA)  ->  LDS  ->
// gate nonlinearity -> c/h update -> h packed for next step.
//
// Block owns a 64-wide cell chunk j0..j0+63 and computes the matching columns
// of ALL FOUR gate sections (i:j, f:H+j, g:2H+j, o:3H+j), so the epilogue is
// block-local.  8 waves: wave = (section 0..3) x (32-col half); each wave does
// a 32(M=batch) x 32 tile over K=1024.  Results staged in 32KB LDS.
// hIn/hOut are ping-pong packed-A buffers (no cross-block hazard).
// ---------------------------------------------------------------------------
__global__ void __launch_bounds__(256)
lstm_step_fused(const __hip_bfloat16* __restrict__ hIn,   // packed A [2][32][32][16]
                const __hip_bfloat16* __restrict__ Whp,   // packed B [256nt][32kt][32][16]
                const float* __restrict__ Gpre,           // [B*T, 4H] input-proj gates
                const float* __restrict__ bsum,           // [4H]
                float* __restrict__ cst,                  // [B, H]
                float* __restrict__ Hbuf,                 // [B*T, H] f32 h outputs
                __hip_bfloat16* __restrict__ hOut,        // packed A for next step
                int t) {
  __shared__ float Gs[4 * 32 * 64];                       // [sec][batch 32][col 64]

  const int lane = threadIdx.x & 31;
  const int wave = threadIdx.x >> 5;
  const int sec  = wave >> 1;             // gate section 0..3
  const int nh   = wave & 1;              // 32-col half of the chunk
  const int j0   = blockIdx.x * 64;       // cell chunk base, 16 blocks cover H
  const int gcol = sec * H_ + j0 + nh * 32;   // global gate column base
  const int Kt   = H_ / 32;               // 32 k-tiles

  const v16bf* A  = (const v16bf*)hIn;
  const v16bf* Bm = (const v16bf*)Whp;

  long aIdx[2], bIdx[2];
#pragma unroll
  for (int i = 0; i < 2; ++i) aIdx[i] = ((long)i * Kt) * 32 + lane;
#pragma unroll
  for (int j = 0; j < 2; ++j) bIdx[j] = ((long)((gcol >> 4) + j) * Kt) * 32 + lane;

  v8f acc[2][2];
#pragma unroll
  for (int i = 0; i < 2; ++i)
#pragma unroll
    for (int j = 0; j < 2; ++j) acc[i][j] = {};

  for (int kt = 0; kt < Kt; ++kt) {
    v16bf a0 = A[aIdx[0]];
    v16bf a1 = A[aIdx[1]];
    v16bf b0 = Bm[bIdx[0]];
    v16bf b1 = Bm[bIdx[1]];
    acc[0][0] = __builtin_amdgcn_wmma_f32_16x16x32_bf16(false, a0, false, b0, (short)0, acc[0][0], false, false);
    acc[0][1] = __builtin_amdgcn_wmma_f32_16x16x32_bf16(false, a0, false, b1, (short)0, acc[0][1], false, false);
    acc[1][0] = __builtin_amdgcn_wmma_f32_16x16x32_bf16(false, a1, false, b0, (short)0, acc[1][0], false, false);
    acc[1][1] = __builtin_amdgcn_wmma_f32_16x16x32_bf16(false, a1, false, b1, (short)0, acc[1][1], false, false);
    aIdx[0] += 32; aIdx[1] += 32; bIdx[0] += 32; bIdx[1] += 32;
  }

  // stage accumulators to LDS: Gs[sec][batch][col-in-chunk]
  const int mrow = (lane >> 4) << 3;
  const int ncol = lane & 15;
#pragma unroll
  for (int i = 0; i < 2; ++i) {
#pragma unroll
    for (int jt = 0; jt < 2; ++jt) {
      int gm  = (i << 4) + mrow;                  // batch row 0..31
      int col = nh * 32 + (jt << 4) + ncol;       // 0..63 within chunk
#pragma unroll
      for (int v = 0; v < 8; ++v)
        Gs[((sec * 32 + gm + v) << 6) + col] = acc[i][jt][v];
    }
  }
  __syncthreads();

  // fused gate epilogue: 32 batch x 64 cells = 2048 cells / 256 threads
#pragma unroll
  for (int e = 0; e < 8; ++e) {
    int cell = threadIdx.x + (e << 8);
    int b  = cell >> 6;
    int jj = cell & 63;
    int j  = j0 + jj;
    long r = (long)b * T_ + t;
    const float* gp = Gpre + r * G4_;
    float gi = Gs[((0 * 32 + b) << 6) + jj] + gp[j]          + bsum[j];
    float gf = Gs[((1 * 32 + b) << 6) + jj] + gp[j + H_]     + bsum[j + H_];
    float gg = Gs[((2 * 32 + b) << 6) + jj] + gp[j + 2 * H_] + bsum[j + 2 * H_];
    float go = Gs[((3 * 32 + b) << 6) + jj] + gp[j + 3 * H_] + bsum[j + 3 * H_];
    float si = 1.0f / (1.0f + __expf(-gi));
    float sf = 1.0f / (1.0f + __expf(-gf));
    float so = 1.0f / (1.0f + __expf(-go));
    int ci = b * H_ + j;
    float c = sf * cst[ci] + si * tanhf(gg);
    float h = so * tanhf(c);
    cst[ci] = c;
    Hbuf[r * H_ + j] = h;
    // scatter into packed A layout for next step's h @ Whh^T
    int mt = b >> 4, mr = b & 15;
    int kt = j >> 5, kk = j & 31;
    int lane2 = mr + (((kk >> 3) & 1) << 4);
    int j16   = (kk & 7) | ((kk >> 4) << 3);
    hOut[((((long)mt * (H_ / 32) + kt) << 5) + lane2) * 16 + j16] = __float2bfloat16(h);
  }
}

__global__ void bias_sum_kernel(const float* __restrict__ a,
                                const float* __restrict__ b,
                                float* __restrict__ o, int n) {
  int i = blockIdx.x * blockDim.x + threadIdx.x;
  if (i < n) o[i] = a[i] + b[i];
}

// ---------------------------------------------------------------------------
// Host-side launchers
// ---------------------------------------------------------------------------
static inline void launch_pack(const float* src, __hip_bfloat16* dst, int R, int K,
                               hipStream_t s) {
  long total = (long)(R / 16) * (K / 32) * 32;
  int blocks = (int)((total + 255) / 256);
  pack_tiles_kernel<<<blocks, 256, 0, s>>>(src, dst, R, K);
}

static inline void launch_gemm(const __hip_bfloat16* A, const __hip_bfloat16* Bp,
                               float* C, const float* bias, int M, int N, int K,
                               hipStream_t s) {
  dim3 grid(N / 256, (M + 63) / 64);
  gemm_bf16_tn<<<grid, 256, 0, s>>>(A, Bp, C, bias, M, N, K);
}

extern "C" void kernel_launch(void* const* d_in, const int* in_sizes, int n_in,
                              void* d_out, int out_size, void* d_ws, size_t ws_size,
                              hipStream_t stream) {
  const float* x    = (const float*)d_in[0];
  const float* Wih0 = (const float*)d_in[1];
  const float* Whh0 = (const float*)d_in[2];
  const float* bih0 = (const float*)d_in[3];
  const float* bhh0 = (const float*)d_in[4];
  const float* Wih1 = (const float*)d_in[5];
  const float* Whh1 = (const float*)d_in[6];
  const float* bih1 = (const float*)d_in[7];
  const float* bhh1 = (const float*)d_in[8];
  const float* Wp   = (const float*)d_in[9];
  const float* bp   = (const float*)d_in[10];

  char* ws = (char*)d_ws;
  size_t off = 0;
  auto alloc = [&](size_t bytes) -> void* {
    off = (off + 255) & ~(size_t)255;
    void* p = ws + off;
    off += bytes;
    return p;
  };

  const long MT = (long)B_ * T_;  // 16384 rows
  __hip_bfloat16* apack = (__hip_bfloat16*)alloc(MT * IN_ * 2);       // x pack / h pack (reused)
  __hip_bfloat16* wih0p = (__hip_bfloat16*)alloc((long)G4_ * IN_ * 2);
  __hip_bfloat16* whh0p = (__hip_bfloat16*)alloc((long)G4_ * H_ * 2);
  __hip_bfloat16* wih1p = (__hip_bfloat16*)alloc((long)G4_ * H_ * 2);
  __hip_bfloat16* whh1p = (__hip_bfloat16*)alloc((long)G4_ * H_ * 2);
  __hip_bfloat16* wpp   = (__hip_bfloat16*)alloc((long)OUT_ * H_ * 2);
  float* bsum0 = (float*)alloc(G4_ * 4);
  float* bsum1 = (float*)alloc(G4_ * 4);
  float* G     = (float*)alloc(MT * (long)G4_ * 4);   // 256 MB input-proj gates (reused per layer)
  float* Hbuf  = (float*)alloc(MT * (long)H_ * 4);    // h outputs of current layer, f32
  const size_t hA_bytes = (size_t)2 * (H_ / 32) * 32 * 16 * 2;   // 64 KB packed A
  __hip_bfloat16* hA0 = (__hip_bfloat16*)alloc(hA_bytes);        // ping-pong h buffers
  __hip_bfloat16* hA1 = (__hip_bfloat16*)alloc(hA_bytes);
  float* cst = (float*)alloc((long)B_ * H_ * 4);

  // 1) pack all static operands (bf16, WMMA tile layout)
  launch_pack(x,    apack, (int)MT, IN_, stream);
  launch_pack(Wih0, wih0p, G4_, IN_, stream);
  launch_pack(Whh0, whh0p, G4_, H_,  stream);
  launch_pack(Wih1, wih1p, G4_, H_,  stream);
  launch_pack(Whh1, whh1p, G4_, H_,  stream);
  launch_pack(Wp,   wpp,   OUT_, H_, stream);
  bias_sum_kernel<<<(G4_ + 255) / 256, 256, 0, stream>>>(bih0, bhh0, bsum0, G4_);
  bias_sum_kernel<<<(G4_ + 255) / 256, 256, 0, stream>>>(bih1, bhh1, bsum1, G4_);

  // 2) layer-0 input projection over all timesteps: G = x @ Wih0^T
  launch_gemm(apack, wih0p, G, nullptr, (int)MT, G4_, IN_, stream);

  for (int layer = 0; layer < 2; ++layer) {
    const __hip_bfloat16* whp = layer ? whh1p : whh0p;
    const float* bs = layer ? bsum1 : bsum0;
    hipMemsetAsync(hA0, 0, hA_bytes, stream);
    hipMemsetAsync(cst, 0, (size_t)B_ * H_ * 4, stream);

    // 3) sequential recurrence: ONE fused kernel per step (WMMA + gates)
    for (int t = 0; t < T_; ++t) {
      const __hip_bfloat16* hin = (t & 1) ? hA1 : hA0;
      __hip_bfloat16*      hout = (t & 1) ? hA0 : hA1;
      lstm_step_fused<<<H_ / 64, 256, 0, stream>>>(hin, whp, G, bs, cst, Hbuf,
                                                   hout, t);
    }

    // 4) lift this layer's h sequence into the next big GEMM
    launch_pack(Hbuf, apack, (int)MT, H_, stream);
    if (layer == 0) {
      // layer-1 input projection: G = h1 @ Wih1^T
      launch_gemm(apack, wih1p, G, nullptr, (int)MT, G4_, H_, stream);
    } else {
      // final projection straight into d_out: out = h2 @ Wp^T + bp
      launch_gemm(apack, wpp, (float*)d_out, bp, (int)MT, OUT_, H_, stream);
    }
  }
}
